// NNConvRegressor_60601988547141
// MI455X (gfx1250) — compile-verified, matrix operand order checked by reading, not run on hardware
//
#include <hip/hip_runtime.h>
#include <hip/hip_bf16.h>

#define NN 10000     // nodes
#define NE 20000     // edges
#define NB 500       // graphs
#define H  64
#define NODE_D 11
#define EDGE_D 5
#define LAYERS 3
#define S2S_STEPS 3
#define BN_EPS 1e-5f

typedef __attribute__((ext_vector_type(16))) __bf16 v16bf;
typedef __attribute__((ext_vector_type(8)))  float  v8f;

// ---------------------------------------------------------------- utilities
__global__ void fill_f32(float* p, float v, int n) {
    int i = blockIdx.x * blockDim.x + threadIdx.x;
    if (i < n) p[i] = v;
}
__global__ void fill_u32(unsigned* p, unsigned v, int n) {
    int i = blockIdx.x * blockDim.x + threadIdx.x;
    if (i < n) p[i] = v;
}

__device__ __forceinline__ float sigmoidf_(float x) { return 1.0f / (1.0f + expf(-x)); }

// ---------------------------------------------------------------- encoder
__global__ void encode_kernel(const float* __restrict__ x, const float* __restrict__ W,
                              const float* __restrict__ b, float* __restrict__ h) {
    int i = blockIdx.x * blockDim.x + threadIdx.x;
    if (i >= NN * H) return;
    int n = i >> 6, j = i & 63;
    float acc = b[j];
#pragma unroll
    for (int d = 0; d < NODE_D; d++) acc += x[n * NODE_D + d] * W[d * H + j];
    h[i] = fmaxf(acc, 0.0f);
}

// ---------------------------------------------------------------- degree
__global__ void deg_accum(const int* __restrict__ ei, float* __restrict__ deg) {
    int e = blockIdx.x * blockDim.x + threadIdx.x;
    if (e < NE) atomicAdd(&deg[ei[NE + e]], 1.0f);
}
__global__ void deg_clamp(float* deg) {
    int n = blockIdx.x * blockDim.x + threadIdx.x;
    if (n < NN) deg[n] = fmaxf(deg[n], 1.0f);
}

// ---------------------------------------------------------------- edge MLP stage 1
__global__ void edge_mlp1(const float* __restrict__ ea, const float* __restrict__ W,
                          const float* __restrict__ b, float* __restrict__ e1) {
    int i = blockIdx.x * blockDim.x + threadIdx.x;
    if (i >= NE * H) return;
    int e = i >> 6, j = i & 63;
    float acc = b[j];
#pragma unroll
    for (int d = 0; d < EDGE_D; d++) acc += ea[e * EDGE_D + d] * W[d * H + j];
    e1[i] = fmaxf(acc, 0.0f);
}

// ------------------------------------------------------- B fragment pre-shuffle
// Bfrag[((tile_n*128 + s)*32 + lane)*16 + j]  (bf16), K index kk = s*32 + kl,
// kl = (j/8)*16 + (lane/16)*8 + j%8 ; kk -> (k_e1 = kk>>6, h = kk&63);
// B[kk][o] = eW2[k_e1][h*64 + o],  o = tile_n*16 + lane%16
__global__ void prep_bfrag(const float* __restrict__ eW2, __bf16* __restrict__ bf) {
    int i = blockIdx.x * blockDim.x + threadIdx.x;
    if (i >= 4 * 128 * 32 * 16) return;
    int j = i & 15;
    int lane = (i >> 4) & 31;
    int s = (i >> 9) & 127;
    int tn = i >> 16;
    int g = lane >> 4, n = lane & 15;
    int kl = ((j >> 3) << 4) + (g << 3) + (j & 7);
    int kk = s * 32 + kl;
    int ke = kk >> 6, hi = kk & 63;
    bf[i] = (__bf16)eW2[ke * (H * H) + hi * H + tn * 16 + n];
}

// ---------------------------------------- fused outer-product GEMM (WMMA bf16)
// msg[E,64] = (e1 outer h_src)[E,4096] @ B[4096,64]
// Block = 4 waves, 64 edges staged in LDS. Each wave computes 16 edges x all
// 64 output cols (4 N-tiles): ONE A-fragment build feeds 4 WMMAs, amortizing
// the packed-bf16 scaling; next slice's 4 B fragments are prefetched.
__global__ void __launch_bounds__(128) nnconv_gemm(const float* __restrict__ hA,
                                                   const float* __restrict__ e1,
                                                   const int* __restrict__ ei,
                                                   const __bf16* __restrict__ bfrag,
                                                   float* __restrict__ msg) {
    __shared__ float hs[64][H + 1];
    __shared__ float es[64][H + 1];
    int tid = threadIdx.x;
    int eb = blockIdx.x * 64;
    for (int t = tid; t < 64 * H; t += 128) {
        int r = t >> 6, c = t & 63;
        int e = eb + r;
        if (e >= NE) e = NE - 1;            // clamp (duplicate rows, unused)
        hs[r][c] = hA[ei[e] * H + c];       // h[src[e]]
        es[r][c] = e1[e * H + c];
    }
    __syncthreads();

    int wave = tid >> 5, lane = tid & 31;
    int g = lane >> 4, m = lane & 15;
    int row = wave * 16 + m;                // edge row within block (this wave)

    // h-part of A fragment (ISA 16-bit A 16x32 layout), packed bf16, per K-parity
    v16bf hb0, hb1;
#pragma unroll
    for (int j = 0; j < 16; j++) {
        int local = ((j >> 3) << 4) + (g << 3) + (j & 7);
        hb0[j] = (__bf16)hs[row][local];
        hb1[j] = (__bf16)hs[row][32 + local];
    }

    const v16bf* bb = ((const v16bf*)bfrag) + lane;
    v8f c0 = {0.f, 0.f, 0.f, 0.f, 0.f, 0.f, 0.f, 0.f};
    v8f c1 = c0, c2 = c0, c3 = c0;

    v16bf bc0 = bb[(0 * 128) * 32];
    v16bf bc1 = bb[(1 * 128) * 32];
    v16bf bc2 = bb[(2 * 128) * 32];
    v16bf bc3 = bb[(3 * 128) * 32];

#pragma unroll 2
    for (int s = 0; s < 128; s++) {
        // prefetch next slice's 4 fragments (s=127 reads slack/neighbor; unused)
        v16bf bn0 = bb[(0 * 128 + s + 1) * 32];
        v16bf bn1 = bb[(1 * 128 + s + 1) * 32];
        v16bf bn2 = bb[(2 * 128 + s + 1) * 32];
        v16bf bn3 = bb[(3 * 128 + s + 1) * 32];
        __bf16 ab = (__bf16)es[row][s >> 1];
        v16bf a = ((s & 1) ? hb1 : hb0) * ab;     // v_pk_mul_bf16 x8, once per 4 WMMAs
        c0 = __builtin_amdgcn_wmma_f32_16x16x32_bf16(false, a, false, bc0, (short)0, c0, false, false);
        c1 = __builtin_amdgcn_wmma_f32_16x16x32_bf16(false, a, false, bc1, (short)0, c1, false, false);
        c2 = __builtin_amdgcn_wmma_f32_16x16x32_bf16(false, a, false, bc2, (short)0, c2, false, false);
        c3 = __builtin_amdgcn_wmma_f32_16x16x32_bf16(false, a, false, bc3, (short)0, c3, false, false);
        bc0 = bn0; bc1 = bn1; bc2 = bn2; bc3 = bn3;
    }

    // C/D layout: lane group g -> rows g*8 + r, col = lane%16
#pragma unroll
    for (int r = 0; r < 8; r++) {
        int er = eb + wave * 16 + g * 8 + r;
        if (er < NE) {
            float* mp = msg + (size_t)er * H + m;
            mp[0]  = c0[r];
            mp[16] = c1[r];
            mp[32] = c2[r];
            mp[48] = c3[r];
        }
    }
}

// ------------------------------------------ bias + scatter-add to destination
__global__ void scatter_msg(const float* __restrict__ msg, const float* __restrict__ hA,
                            const int* __restrict__ ei, const float* __restrict__ eb2,
                            float* __restrict__ agg) {
    int i = blockIdx.x * blockDim.x + threadIdx.x;
    if (i >= NE * H) return;
    int e = i >> 6, o = i & 63;
    int sn = ei[e], dn = ei[NE + e];
    const float* hr = hA + (size_t)sn * H;
    float bias = 0.0f;
#pragma unroll 8
    for (int h2 = 0; h2 < H; h2++) bias += hr[h2] * eb2[h2 * H + o];
    atomicAdd(&agg[(size_t)dn * H + o], msg[i] + bias);
}

// ---------------------------------------------------------------- root update
__global__ void root_update(const float* __restrict__ agg, const float* __restrict__ deg,
                            const float* __restrict__ hA, const float* __restrict__ root,
                            const float* __restrict__ cbias, float* __restrict__ hN) {
    int i = blockIdx.x * blockDim.x + threadIdx.x;
    if (i >= NN * H) return;
    int n = i >> 6, o = i & 63;
    const float* hr = hA + (size_t)n * H;
    float acc = cbias[o] + agg[i] / deg[n];
#pragma unroll 8
    for (int h2 = 0; h2 < H; h2++) acc += hr[h2] * root[h2 * H + o];
    hN[i] = acc;
}

// ---------------------------------------------------------------- batchnorm
__global__ void bn_stats(const float* __restrict__ hN, float* __restrict__ stats) {
    int c = blockIdx.x;           // channel
    int t = threadIdx.x;
    float s = 0.f, s2 = 0.f;
    for (int n = t; n < NN; n += blockDim.x) {
        float v = hN[(size_t)n * H + c];
        s += v; s2 += v * v;
    }
    __shared__ float rs[256], rq[256];
    rs[t] = s; rq[t] = s2;
    __syncthreads();
    for (int off = 128; off > 0; off >>= 1) {
        if (t < off) { rs[t] += rs[t + off]; rq[t] += rq[t + off]; }
        __syncthreads();
    }
    if (t == 0) { stats[c] = rs[0]; stats[H + c] = rq[0]; }
}

__global__ void bn_apply(const float* __restrict__ hN, const float* __restrict__ stats,
                         const float* __restrict__ gamma, const float* __restrict__ beta,
                         float* __restrict__ hA) {
    int i = blockIdx.x * blockDim.x + threadIdx.x;
    if (i >= NN * H) return;
    int o = i & 63;
    const float inv_n = 1.0f / (float)NN;
    float mu = stats[o] * inv_n;
    float var = stats[H + o] * inv_n - mu * mu;
    float v = (hN[i] - mu) * rsqrtf(var + BN_EPS) * gamma[o] + beta[o];
    hA[i] = fmaxf(v, 0.0f) + hA[i];     // relu + residual
}

// ---------------------------------------------------------------- Set2Set
__global__ void lstm_gates(const float* __restrict__ qstar, const float* __restrict__ hl,
                           const float* __restrict__ Wih, const float* __restrict__ Whh,
                           const float* __restrict__ bih, const float* __restrict__ bhh,
                           float* __restrict__ gates) {
    int i = blockIdx.x * blockDim.x + threadIdx.x;
    if (i >= NB * 4 * H) return;
    int b = i >> 8, j = i & 255;
    const float* q = qstar + (size_t)b * 2 * H;
    const float* hh = hl + (size_t)b * H;
    float acc = bih[j] + bhh[j];
#pragma unroll 8
    for (int k = 0; k < 2 * H; k++) acc += q[k] * Wih[j * 2 * H + k];
#pragma unroll 8
    for (int k = 0; k < H; k++) acc += hh[k] * Whh[j * H + k];
    gates[i] = acc;
}

__global__ void lstm_update(const float* __restrict__ gates, float* __restrict__ hl,
                            float* __restrict__ cl) {
    int i = blockIdx.x * blockDim.x + threadIdx.x;
    if (i >= NB * H) return;
    int b = i >> 6, j = i & 63;
    const float* g = gates + (size_t)b * 4 * H;
    float ig = sigmoidf_(g[j]);
    float fg = sigmoidf_(g[H + j]);
    float gg = tanhf(g[2 * H + j]);
    float og = sigmoidf_(g[3 * H + j]);
    float c = fg * cl[i] + ig * gg;
    cl[i] = c;
    hl[i] = og * tanhf(c);
}

__device__ __forceinline__ unsigned f32_key(float f) {
    unsigned u = __float_as_uint(f);
    return (u & 0x80000000u) ? ~u : (u | 0x80000000u);
}
__device__ __forceinline__ float key_f32(unsigned k) {
    unsigned u = (k & 0x80000000u) ? (k ^ 0x80000000u) : ~k;
    return __uint_as_float(u);
}

__global__ void attn_score(const float* __restrict__ hA, const float* __restrict__ hl,
                           const int* __restrict__ batch, float* __restrict__ esc,
                           unsigned* __restrict__ mkey) {
    int n = blockIdx.x * blockDim.x + threadIdx.x;
    if (n >= NN) return;
    int b = batch[n];
    const float* hr = hA + (size_t)n * H;
    const float* q = hl + (size_t)b * H;
    float acc = 0.0f;
#pragma unroll 8
    for (int j = 0; j < H; j++) acc += hr[j] * q[j];
    esc[n] = acc;
    atomicMax(&mkey[b], f32_key(acc));
}

__global__ void attn_exp(float* __restrict__ esc, const int* __restrict__ batch,
                         const unsigned* __restrict__ mkey, float* __restrict__ sden) {
    int n = blockIdx.x * blockDim.x + threadIdx.x;
    if (n >= NN) return;
    int b = batch[n];
    float ex = expf(esc[n] - key_f32(mkey[b]));
    esc[n] = ex;
    atomicAdd(&sden[b], ex);
}

__global__ void attn_weighted(const float* __restrict__ esc, const float* __restrict__ sden,
                              const int* __restrict__ batch, const float* __restrict__ hA,
                              float* __restrict__ r) {
    int i = blockIdx.x * blockDim.x + threadIdx.x;
    if (i >= NN * H) return;
    int n = i >> 6, o = i & 63;
    int b = batch[n];
    float a = esc[n] / sden[b];
    atomicAdd(&r[(size_t)b * H + o], a * hA[i]);
}

__global__ void qstar_update(const float* __restrict__ hl, const float* __restrict__ r,
                             float* __restrict__ qstar) {
    int i = blockIdx.x * blockDim.x + threadIdx.x;
    if (i >= NB * 2 * H) return;
    int b = i >> 7, j = i & 127;
    qstar[i] = (j < H) ? hl[(size_t)b * H + j] : r[(size_t)b * H + (j - H)];
}

// ---------------------------------------------------------------- head
__global__ void __launch_bounds__(64) head_kernel(const float* __restrict__ qstar,
                                                  const float* __restrict__ hW1, const float* __restrict__ hb1,
                                                  const float* __restrict__ hW2, const float* __restrict__ hb2,
                                                  const float* __restrict__ hW3, const float* __restrict__ hb3,
                                                  float* __restrict__ out) {
    int b = blockIdx.x, t = threadIdx.x;
    __shared__ float z1[H];
    __shared__ float z2[H / 2];
    const float* q = qstar + (size_t)b * 2 * H;
    float acc = hb1[t];
#pragma unroll 8
    for (int k = 0; k < 2 * H; k++) acc += q[k] * hW1[k * H + t];
    z1[t] = fmaxf(acc, 0.0f);
    __syncthreads();
    if (t < H / 2) {
        float a2 = hb2[t];
#pragma unroll 8
        for (int k = 0; k < H; k++) a2 += z1[k] * hW2[k * (H / 2) + t];
        z2[t] = fmaxf(a2, 0.0f);
    }
    __syncthreads();
    if (t == 0) {
        float a3 = hb3[0];
#pragma unroll
        for (int k = 0; k < H / 2; k++) a3 += z2[k] * hW3[k];
        out[b] = a3;
    }
}

// ---------------------------------------------------------------- launch
extern "C" void kernel_launch(void* const* d_in, const int* in_sizes, int n_in,
                              void* d_out, int out_size, void* d_ws, size_t ws_size,
                              hipStream_t stream) {
    (void)in_sizes; (void)n_in; (void)out_size; (void)ws_size;
    const float* x     = (const float*)d_in[0];
    const int*   ei    = (const int*)d_in[1];
    const float* ea    = (const float*)d_in[2];
    const int*   batch = (const int*)d_in[3];
    const float* W_enc = (const float*)d_in[4];
    const float* b_enc = (const float*)d_in[5];
    const float* eW1   = (const float*)d_in[6];
    const float* eb1   = (const float*)d_in[7];
    const float* eW2   = (const float*)d_in[8];
    const float* eb2   = (const float*)d_in[9];
    const float* root  = (const float*)d_in[10];
    const float* cbias = (const float*)d_in[11];
    const float* gamma = (const float*)d_in[12];
    const float* beta  = (const float*)d_in[13];
    const float* Wih   = (const float*)d_in[14];
    const float* Whh   = (const float*)d_in[15];
    const float* bih   = (const float*)d_in[16];
    const float* bhh   = (const float*)d_in[17];
    const float* hW1   = (const float*)d_in[18];
    const float* hb1   = (const float*)d_in[19];
    const float* hW2   = (const float*)d_in[20];
    const float* hb2   = (const float*)d_in[21];
    const float* hW3   = (const float*)d_in[22];
    const float* hb3   = (const float*)d_in[23];
    float* out = (float*)d_out;

    char* ws = (char*)d_ws;
    auto carve = [&](size_t bytes) -> char* {
        char* p = ws;
        ws += (bytes + 255) & ~(size_t)255;
        return p;
    };
    float*    hA    = (float*)carve((size_t)NN * H * 4);
    float*    hN    = (float*)carve((size_t)NN * H * 4);
    float*    e1b   = (float*)carve((size_t)NE * H * 4);
    float*    msg   = (float*)carve((size_t)NE * H * 4);
    float*    agg   = (float*)carve((size_t)NN * H * 4);
    float*    deg   = (float*)carve((size_t)NN * 4);
    __bf16*   bfrag = (__bf16*)carve((size_t)(4 * 128 + 2) * 32 * 16 * 2); // +2 slices slack for prefetch
    float*    stats = (float*)carve(2 * H * 4);
    float*    gates = (float*)carve((size_t)NB * 4 * H * 4);
    float*    qstar = (float*)carve((size_t)NB * 2 * H * 4);
    float*    hl    = (float*)carve((size_t)NB * H * 4);
    float*    cl    = (float*)carve((size_t)NB * H * 4);
    float*    rbuf  = (float*)carve((size_t)NB * H * 4);
    float*    esc   = (float*)carve((size_t)NN * 4);
    unsigned* mkey  = (unsigned*)carve((size_t)NB * 4);
    float*    sden  = (float*)carve((size_t)NB * 4);

#define GRID1(n) dim3(((n) + 255) / 256), dim3(256), 0, stream

    encode_kernel<<<GRID1(NN * H)>>>(x, W_enc, b_enc, hA);
    fill_f32<<<GRID1(NN)>>>(deg, 0.0f, NN);
    deg_accum<<<GRID1(NE)>>>(ei, deg);
    deg_clamp<<<GRID1(NN)>>>(deg);

    for (int l = 0; l < LAYERS; l++) {
        edge_mlp1<<<GRID1(NE * H)>>>(ea, eW1 + l * EDGE_D * H, eb1 + l * H, e1b);
        prep_bfrag<<<GRID1(4 * 128 * 32 * 16)>>>(eW2 + (size_t)l * H * H * H, bfrag);
        fill_f32<<<GRID1(NN * H)>>>(agg, 0.0f, NN * H);
        nnconv_gemm<<<dim3((NE + 63) / 64), dim3(128), 0, stream>>>(hA, e1b, ei, bfrag, msg);
        scatter_msg<<<GRID1(NE * H)>>>(msg, hA, ei, eb2 + (size_t)l * H * H, agg);
        root_update<<<GRID1(NN * H)>>>(agg, deg, hA, root + (size_t)l * H * H, cbias + l * H, hN);
        bn_stats<<<dim3(H), dim3(256), 0, stream>>>(hN, stats);
        bn_apply<<<GRID1(NN * H)>>>(hN, stats, gamma + l * H, beta + l * H, hA);
    }

    fill_f32<<<GRID1(NB * 2 * H)>>>(qstar, 0.0f, NB * 2 * H);
    fill_f32<<<GRID1(NB * H)>>>(hl, 0.0f, NB * H);
    fill_f32<<<GRID1(NB * H)>>>(cl, 0.0f, NB * H);

    for (int s = 0; s < S2S_STEPS; s++) {
        lstm_gates<<<GRID1(NB * 4 * H)>>>(qstar, hl, Wih, Whh, bih, bhh, gates);
        lstm_update<<<GRID1(NB * H)>>>(gates, hl, cl);
        fill_u32<<<GRID1(NB)>>>(mkey, 0u, NB);
        attn_score<<<GRID1(NN)>>>(hA, hl, batch, esc, mkey);
        fill_f32<<<GRID1(NB)>>>(sden, 0.0f, NB);
        fill_f32<<<GRID1(NB * H)>>>(rbuf, 0.0f, NB * H);
        attn_exp<<<GRID1(NN)>>>(esc, batch, mkey, sden);
        attn_weighted<<<GRID1(NN * H)>>>(esc, sden, batch, hA, rbuf);
        qstar_update<<<GRID1(NB * 2 * H)>>>(hl, rbuf, qstar);
    }

    head_kernel<<<dim3(NB), dim3(64), 0, stream>>>(qstar, hW1, hb1, hW2, hb2, hW3, hb3, out);
#undef GRID1
}
